// MultiHeadAttention_40870908789096
// MI455X (gfx1250) — matrix operand order repaired and checked by reading, not compiled, not run on hardware
//
#include <hip/hip_runtime.h>
#include <stdint.h>

// ---------------------------------------------------------------------------
// MI455X (gfx1250) multi-head attention with 2D log-bucket relative bias.
// bf16 WMMA (v_wmma_f32_16x16x32_bf16) everywhere a GEMM appears.
// GEMM is register-blocked 32x64 per wave (8 WMMAs per K=32 step).
// ---------------------------------------------------------------------------

typedef __attribute__((ext_vector_type(16))) __bf16        v16bf;
typedef __attribute__((ext_vector_type(8)))  float         v8f;
typedef __attribute__((ext_vector_type(4)))  unsigned int  v4u;

#define B_   4
#define G_   1024
#define D_   512
#define H_   8
#define DH_  64
#define NBUCK 32

__device__ __forceinline__ unsigned short f32_to_bf16_bits(float f) {
  unsigned int u = __float_as_uint(f);
  u += 0x7FFFu + ((u >> 16) & 1u);          // round to nearest even
  return (unsigned short)(u >> 16);
}

// A/B operand container: 16 bf16 per lane = 8 VGPRs, loaded as two b128s.
union ABf {
  v16bf v;
  v4u   q[2];
  unsigned short s[16];
};

__device__ __forceinline__ v4u ld128(const __bf16* p) {
  return *reinterpret_cast<const v4u*>(p);
}

__device__ __forceinline__ int rel_bucket(float delta) {
  float s = (delta > 0.f) ? 1.f : ((delta < 0.f) ? -1.f : 0.f);
  float n = fminf(fmaxf(fabsf(delta), 1e-6f), 128.f);
  int li = (int)floorf(__log2f(n));
  li = li < 0 ? 0 : (li > (NBUCK - 1) ? (NBUCK - 1) : li);
  return li * (int)s + (NBUCK - 1);
}

// ---------------------------------------------------------------------------
// f32 -> bf16 bit conversion
// ---------------------------------------------------------------------------
__global__ void k_f32_to_bf16(const float* __restrict__ src,
                              unsigned short* __restrict__ dst, int n) {
  int i = blockIdx.x * 256 + threadIdx.x;
  if (i < n) dst[i] = f32_to_bf16_bits(src[i]);
}

// ---------------------------------------------------------------------------
// Precompute packed (bucket_x | bucket_y<<8) per (b, q, k) pair.
// Head-independent: saves 8x transcendental work vs in-attention compute.
// ---------------------------------------------------------------------------
__global__ void k_buckets(const float* __restrict__ coords,
                          unsigned short* __restrict__ bk) {
  long long i = (long long)blockIdx.x * 256 + threadIdx.x;
  if (i >= (long long)B_ * G_ * G_) return;
  int k = (int)(i % G_);
  long long t = i / G_;
  int q = (int)(t % G_);
  int b = (int)(t / G_);
  const float* cq = coords + ((size_t)b * G_ + q) * 2;
  const float* ck = coords + ((size_t)b * G_ + k) * 2;
  int bx = rel_bucket(cq[0] - ck[0]);
  int by = rel_bucket(cq[1] - ck[1]);
  bk[i] = (unsigned short)(bx | (by << 8));
}

// ---------------------------------------------------------------------------
// WMMA GEMM:  C[m,n] = sum_k A[m,k]*W[n,k] + bias[n]
//   A: M x K bf16 row-major, W: N x K bf16 row-major (torch y = x @ W^T).
//   Register-blocked: one wave owns a 32(M) x 64(N) block = 2x4 subtiles.
//   Per K=32 step: 4 A-loads + 8 B-loads feed 8 WMMAs (1.5 b128/WMMA).
//   Block = 128 threads (4 waves, split along M). grid = (M/128, N/64).
// mode 0: write bf16 into (B,H,G,dh)       [Q / K heads]
// mode 2: write bf16 into (B,H,dh,G)       [V transposed for attn*V WMMA]
// mode 3: write f32 row-major M x N        [final output projection]
// ---------------------------------------------------------------------------
__global__ void __launch_bounds__(128)
k_gemm_wmma(const unsigned short* __restrict__ Abits,
            const unsigned short* __restrict__ Wbits,
            const float* __restrict__ bias,
            void* __restrict__ dst,
            int M, int N, int K, int mode) {
  const __bf16* A = reinterpret_cast<const __bf16*>(Abits);
  const __bf16* W = reinterpret_cast<const __bf16*>(Wbits);
  int lane = threadIdx.x & 31, wv = threadIdx.x >> 5;
  int m0 = (blockIdx.x * 4 + wv) * 32;   // wave's first output row
  int n0 = blockIdx.y * 64;              // wave's first output col
  int hf = lane >> 4, l = lane & 15;

  // A rows for the two M-subtiles (both lane halves read same row, split K)
  const __bf16* ap0 = A + (size_t)(m0 + l) * K;
  const __bf16* ap1 = A + (size_t)(m0 + 16 + l) * K;
  const __bf16* wp  = W + (size_t)(n0 + l) * K;   // + j*16*K per N-subtile

  v8f acc[2][4];
#pragma unroll
  for (int i = 0; i < 2; ++i)
#pragma unroll
    for (int j = 0; j < 4; ++j)
#pragma unroll
      for (int r = 0; r < 8; ++r) acc[i][j][r] = 0.f;

  for (int k0 = 0; k0 < K; k0 += 32) {
    ABf a0, a1;
    a0.q[0] = ld128(ap0 + k0 + hf * 8);        // A elems 0-7 : K = hf*8 + e
    a0.q[1] = ld128(ap0 + k0 + 16 + hf * 8);   // A elems 8-15: K = 16+hf*8+e
    a1.q[0] = ld128(ap1 + k0 + hf * 8);
    a1.q[1] = ld128(ap1 + k0 + 16 + hf * 8);
#pragma unroll
    for (int j = 0; j < 4; ++j) {
      ABf b;
      const __bf16* wpj = wp + (size_t)j * 16 * K;
      b.q[0] = ld128(wpj + k0 + hf * 8);       // B elems 0-7 : K = hf*8 + e
      b.q[1] = ld128(wpj + k0 + 16 + hf * 8);
      acc[0][j] = __builtin_amdgcn_wmma_f32_16x16x32_bf16(false, a0.v, false, b.v,
                                                          (short)0, acc[0][j], false, false);
      acc[1][j] = __builtin_amdgcn_wmma_f32_16x16x32_bf16(false, a1.v, false, b.v,
                                                          (short)0, acc[1][j], false, false);
    }
  }

  // C layout: VGPR r -> row m = hf*8 + r, col n = l (within each subtile)
#pragma unroll
  for (int i = 0; i < 2; ++i) {
#pragma unroll
    for (int j = 0; j < 4; ++j) {
#pragma unroll
      for (int r = 0; r < 8; ++r) {
        int m = m0 + i * 16 + hf * 8 + r;
        int n = n0 + j * 16 + l;
        float val = acc[i][j][r] + bias[n];
        if (mode == 3) {
          ((float*)dst)[(size_t)m * N + n] = val;
        } else {
          int b = m >> 10, g = m & (G_ - 1);
          int hh = n >> 6, dd = n & (DH_ - 1);
          size_t idx;
          if (mode == 2)
            idx = (((size_t)b * H_ + hh) * DH_ + dd) * G_ + g;   // (B,H,dh,G)
          else
            idx = (((size_t)b * H_ + hh) * G_ + g) * DH_ + dd;   // (B,H,G,dh)
          ((unsigned short*)dst)[idx] = f32_to_bf16_bits(val);
        }
      }
    }
  }
}

// ---------------------------------------------------------------------------
// Flash attention: one wave owns a 16-row Q tile, streams 32-key tiles.
//   grid = (B*H, G/64), block = 128 (4 waves, 4 q-tiles per block).
// ---------------------------------------------------------------------------
__global__ void __launch_bounds__(128)
k_attn(const unsigned short* __restrict__ Qbits,
       const unsigned short* __restrict__ Kbits,
       const unsigned short* __restrict__ Vtbits,
       const unsigned short* __restrict__ bpack,
       const int* __restrict__ mask,
       const float* __restrict__ rpe_x,
       const float* __restrict__ rpe_y,
       unsigned short* __restrict__ outb /* (B,G,D) bf16 */) {
  __shared__ float sP[4][16][33];                 // per-wave P transpose buffer
  __shared__ float s_rx[(2 * NBUCK - 1) * H_];
  __shared__ float s_ry[(2 * NBUCK - 1) * H_];

  int tid = threadIdx.x;
  for (int i = tid; i < (2 * NBUCK - 1) * H_; i += 128) {
    s_rx[i] = rpe_x[i];
    s_ry[i] = rpe_y[i];
  }
  __syncthreads();

  int lane = tid & 31, wv = tid >> 5;
  int bh = blockIdx.x, b = bh >> 3, h = bh & 7;
  int qt = blockIdx.y * 4 + wv;                   // q-tile index, 0..63
  int hf = lane >> 4, l = lane & 15;

  const __bf16* Q  = reinterpret_cast<const __bf16*>(Qbits)  + ((size_t)bh * G_ + qt * 16) * DH_;
  const __bf16* Kh = reinterpret_cast<const __bf16*>(Kbits)  + (size_t)bh * G_ * DH_;
  const __bf16* Vt = reinterpret_cast<const __bf16*>(Vtbits) + (size_t)bh * DH_ * G_;

  // Q tile in A layout, two K=32 chunks over d_head=64 (held for whole pass)
  ABf qa0, qa1;
  qa0.q[0] = ld128(Q + l * DH_ + hf * 8);
  qa0.q[1] = ld128(Q + l * DH_ + 16 + hf * 8);
  qa1.q[0] = ld128(Q + l * DH_ + 32 + hf * 8);
  qa1.q[1] = ld128(Q + l * DH_ + 32 + 16 + hf * 8);

  const unsigned short* bp_base = bpack + ((size_t)b * G_ + qt * 16) * G_;

  float rmax[8], rsum[8];
  v8f oacc[4];
#pragma unroll
  for (int r = 0; r < 8; ++r) {
    rmax[r] = -INFINITY;
    rsum[r] = 0.f;
    oacc[0][r] = 0.f; oacc[1][r] = 0.f; oacc[2][r] = 0.f; oacc[3][r] = 0.f;
  }

  for (int kt = 0; kt < G_ / 32; ++kt) {
    int kb = kt * 32;

    // --- S = Q K^T for 32 keys (two 16-col tiles, K accumulated over d=64)
    v8f s0, s1;
#pragma unroll
    for (int r = 0; r < 8; ++r) { s0[r] = 0.f; s1[r] = 0.f; }
    {
      ABf kb0, kb1;
      kb0.q[0] = ld128(Kh + (size_t)(kb + l) * DH_ + hf * 8);
      kb0.q[1] = ld128(Kh + (size_t)(kb + l) * DH_ + 16 + hf * 8);
      kb1.q[0] = ld128(Kh + (size_t)(kb + 16 + l) * DH_ + hf * 8);
      kb1.q[1] = ld128(Kh + (size_t)(kb + 16 + l) * DH_ + 16 + hf * 8);
      s0 = __builtin_amdgcn_wmma_f32_16x16x32_bf16(false, qa0.v, false, kb0.v, (short)0, s0, false, false);
      s1 = __builtin_amdgcn_wmma_f32_16x16x32_bf16(false, qa0.v, false, kb1.v, (short)0, s1, false, false);
      kb0.q[0] = ld128(Kh + (size_t)(kb + l) * DH_ + 32 + hf * 8);
      kb0.q[1] = ld128(Kh + (size_t)(kb + l) * DH_ + 32 + 16 + hf * 8);
      kb1.q[0] = ld128(Kh + (size_t)(kb + 16 + l) * DH_ + 32 + hf * 8);
      kb1.q[1] = ld128(Kh + (size_t)(kb + 16 + l) * DH_ + 32 + 16 + hf * 8);
      s0 = __builtin_amdgcn_wmma_f32_16x16x32_bf16(false, qa1.v, false, kb0.v, (short)0, s0, false, false);
      s1 = __builtin_amdgcn_wmma_f32_16x16x32_bf16(false, qa1.v, false, kb1.v, (short)0, s1, false, false);
    }

    // --- scale + relative-position bias + mask (C layout: row hf*8+r, col l)
    int k0i = kb + l, k1i = kb + 16 + l;
    int mk0 = mask[b * G_ + k0i], mk1 = mask[b * G_ + k1i];
#pragma unroll
    for (int r = 0; r < 8; ++r) {
      const unsigned short* row = bp_base + (size_t)(hf * 8 + r) * G_;
      unsigned short p0 = row[k0i], p1 = row[k1i];
      float bias0 = s_rx[(p0 & 255) * H_ + h] + s_ry[(p0 >> 8) * H_ + h];
      float bias1 = s_rx[(p1 & 255) * H_ + h] + s_ry[(p1 >> 8) * H_ + h];
      s0[r] = mk0 ? (s0[r] * 0.125f + bias0) : -INFINITY;
      s1[r] = mk1 ? (s1[r] * 0.125f + bias1) : -INFINITY;
    }

    // --- online softmax (row lives across 16 lanes of this half-group)
#pragma unroll
    for (int r = 0; r < 8; ++r) {
      float v = fmaxf(s0[r], s1[r]);
      for (int off = 1; off < 16; off <<= 1) v = fmaxf(v, __shfl_xor(v, off, 32));
      float nm = fmaxf(rmax[r], v);
      float cf, p0, p1;
      if (nm == -INFINITY) {
        cf = 1.f; p0 = 0.f; p1 = 0.f;
      } else {
        cf = __expf(rmax[r] - nm);
        p0 = __expf(s0[r] - nm);
        p1 = __expf(s1[r] - nm);
      }
      rmax[r] = nm;
      s0[r] = p0; s1[r] = p1;
      float ps = p0 + p1;
      for (int off = 1; off < 16; off <<= 1) ps += __shfl_xor(ps, off, 32);
      rsum[r] = rsum[r] * cf + ps;
      oacc[0][r] *= cf; oacc[1][r] *= cf; oacc[2][r] *= cf; oacc[3][r] *= cf;
    }

    // --- transpose P (C layout -> A layout) through per-wave LDS slice
#pragma unroll
    for (int r = 0; r < 8; ++r) {
      sP[wv][hf * 8 + r][l]      = s0[r];
      sP[wv][hf * 8 + r][16 + l] = s1[r];
    }
    ABf pa;   // A layout, row m = l, K = key-in-tile
#pragma unroll
    for (int e = 0; e < 8; ++e) pa.s[e]     = f32_to_bf16_bits(sP[wv][l][hf * 8 + e]);
#pragma unroll
    for (int e = 0; e < 8; ++e) pa.s[8 + e] = f32_to_bf16_bits(sP[wv][l][16 + hf * 8 + e]);

    // --- O += P @ V   (V stored transposed: lane d-column reads contiguous k)
#pragma unroll
    for (int t = 0; t < 4; ++t) {
      ABf vb;
      vb.q[0] = ld128(Vt + (size_t)(t * 16 + l) * G_ + kb + hf * 8);
      vb.q[1] = ld128(Vt + (size_t)(t * 16 + l) * G_ + kb + 16 + hf * 8);
      oacc[t] = __builtin_amdgcn_wmma_f32_16x16x32_bf16(false, pa.v, false, vb.v,
                                                        (short)0, oacc[t], false, false);
    }
  }

  // --- normalize and write (B,G,D) bf16 for the output projection
#pragma unroll
  for (int t = 0; t < 4; ++t) {
#pragma unroll
    for (int r = 0; r < 8; ++r) {
      float val = oacc[t][r] / rsum[r];
      int q = qt * 16 + hf * 8 + r;
      int n = h * DH_ + t * 16 + l;
      outb[((size_t)b * G_ + q) * D_ + n] = f32_to_bf16_bits(val);
    }
  }
}

// ---------------------------------------------------------------------------
// Host-side orchestration
// ---------------------------------------------------------------------------
extern "C" void kernel_launch(void* const* d_in, const int* in_sizes, int n_in,
                              void* d_out, int out_size, void* d_ws, size_t ws_size,
                              hipStream_t stream) {
  const float* query  = (const float*)d_in[0];
  const float* key    = (const float*)d_in[1];
  const float* value  = (const float*)d_in[2];
  const float* coords = (const float*)d_in[3];
  const int*   mask   = (const int*)d_in[4];
  const float* Wq = (const float*)d_in[5];
  const float* bq = (const float*)d_in[6];
  const float* Wk = (const float*)d_in[7];
  const float* bk = (const float*)d_in[8];
  const float* Wv = (const float*)d_in[9];
  const float* bv = (const float*)d_in[10];
  const float* Wo = (const float*)d_in[11];
  const float* bo = (const float*)d_in[12];
  const float* rpe_x = (const float*)d_in[13];
  const float* rpe_y = (const float*)d_in[14];

  char* ws = (char*)d_ws;
  size_t off = 0;
  auto alloc = [&](size_t bytes) {
    size_t r = off;
    off += (bytes + 255) & ~(size_t)255;
    return r;
  };

  const size_t MG = (size_t)B_ * G_;   // 4096 rows
  unsigned short* q_bf  = (unsigned short*)(ws + alloc(MG * D_ * 2));
  unsigned short* k_bf  = (unsigned short*)(ws + alloc(MG * D_ * 2));
  unsigned short* v_bf  = (unsigned short*)(ws + alloc(MG * D_ * 2));
  unsigned short* wq_bf = (unsigned short*)(ws + alloc((size_t)D_ * D_ * 2));
  unsigned short* wk_bf = (unsigned short*)(ws + alloc((size_t)D_ * D_ * 2));
  unsigned short* wv_bf = (unsigned short*)(ws + alloc((size_t)D_ * D_ * 2));
  unsigned short* wo_bf = (unsigned short*)(ws + alloc((size_t)D_ * D_ * 2));
  unsigned short* Qh    = (unsigned short*)(ws + alloc(MG * D_ * 2));
  unsigned short* Kh    = (unsigned short*)(ws + alloc(MG * D_ * 2));
  unsigned short* Vth   = (unsigned short*)(ws + alloc(MG * D_ * 2));
  // Region reuse (stream-ordered, so safe):
  unsigned short* bpack = k_bf;   // 8 MB bucket table overlays dead k_bf+v_bf staging
  unsigned short* attnb = q_bf;   // attention output overlays dead q_bf staging

  int na = (int)(MG * D_);        // 2,097,152
  int nw = D_ * D_;               // 262,144
  k_f32_to_bf16<<<(na + 255) / 256, 256, 0, stream>>>(query, q_bf, na);
  k_f32_to_bf16<<<(na + 255) / 256, 256, 0, stream>>>(key,   k_bf, na);
  k_f32_to_bf16<<<(na + 255) / 256, 256, 0, stream>>>(value, v_bf, na);
  k_f32_to_bf16<<<(nw + 255) / 256, 256, 0, stream>>>(Wq, wq_bf, nw);
  k_f32_to_bf16<<<(nw + 255) / 256, 256, 0, stream>>>(Wk, wk_bf, nw);
  k_f32_to_bf16<<<(nw + 255) / 256, 256, 0, stream>>>(Wv, wv_bf, nw);
  k_f32_to_bf16<<<(nw + 255) / 256, 256, 0, stream>>>(Wo, wo_bf, nw);

  dim3 gblk(128);
  dim3 ggrid((unsigned)(MG / 128), D_ / 64);   // 32 x 8 blocks, 4 waves/block
  k_gemm_wmma<<<ggrid, gblk, 0, stream>>>(q_bf, wq_bf, bq, Qh,  (int)MG, D_, D_, 0);
  k_gemm_wmma<<<ggrid, gblk, 0, stream>>>(k_bf, wk_bf, bk, Kh,  (int)MG, D_, D_, 0);
  k_gemm_wmma<<<ggrid, gblk, 0, stream>>>(v_bf, wv_bf, bv, Vth, (int)MG, D_, D_, 2);

  long long npair = (long long)B_ * G_ * G_;
  k_buckets<<<(unsigned)((npair + 255) / 256), 256, 0, stream>>>(coords, bpack);

  k_attn<<<dim3(B_ * H_, G_ / 64), 128, 0, stream>>>(Qh, Kh, Vth, bpack, mask,
                                                     rpe_x, rpe_y, attnb);

  k_gemm_wmma<<<ggrid, gblk, 0, stream>>>(attnb, wo_bf, bo, d_out, (int)MG, D_, D_, 3);
}
